// GAT_80015240724626
// MI455X (gfx1250) — compile-verified, matrix operand order checked by reading, not run on hardware
//
#include <hip/hip_runtime.h>
#include <hip/hip_fp16.h>

#define N_NODES 50000
#define N_EDGES 800000
#define HIDDEN  64
#define NHEAD   3
#define F1      192   // NHEAD*HIDDEN
#define NEG_SLOPE 0.2f

typedef __attribute__((ext_vector_type(16))) _Float16 v16h;
typedef __attribute__((ext_vector_type(8)))  _Float16 v8h;
typedef __attribute__((ext_vector_type(8)))  float    v8f;

__device__ __forceinline__ unsigned encF(float f){
    unsigned u = __float_as_uint(f);
    return (u & 0x80000000u) ? ~u : (u | 0x80000000u);
}
__device__ __forceinline__ float decF(unsigned u){
    return __uint_as_float((u & 0x80000000u) ? (u & 0x7FFFFFFFu) : ~u);
}
__device__ __forceinline__ float lrelu(float v){ return v >= 0.f ? v : NEG_SLOPE * v; }
__device__ __forceinline__ float eluf(float v){ return v > 0.f ? v : expm1f(v); }

// ---- precompute fold vectors: u_src[k][h]=sum_c W1s[k,h*64+c]*att1s[h,c]; ce[h]=sum_c W1e[h*64+c]*att1e[h,c]
__global__ void fold_kernel(const float* __restrict__ W1s, const float* __restrict__ W1d,
                            const float* __restrict__ a1s, const float* __restrict__ a1d,
                            const float* __restrict__ W1e, const float* __restrict__ a1e,
                            float* __restrict__ u_src, float* __restrict__ u_dst, float* __restrict__ ce){
    int t = threadIdx.x;            // 0..191
    if (t < 192){
        int k = t / 3, h = t % 3;
        float ss = 0.f, sd = 0.f;
        for (int c = 0; c < HIDDEN; ++c){
            ss += W1s[k*F1 + h*HIDDEN + c] * a1s[h*HIDDEN + c];
            sd += W1d[k*F1 + h*HIDDEN + c] * a1d[h*HIDDEN + c];
        }
        u_src[k*3 + h] = ss;
        u_dst[k*3 + h] = sd;
    }
    if (t < 3){
        float s = 0.f;
        for (int c = 0; c < HIDDEN; ++c) s += W1e[t*HIDDEN + c] * a1e[t*HIDDEN + c];
        ce[t] = s;
    }
}

// ---- pre-pack B (W1_src, 64x192 f32) into WMMA fragment layout (f16), per (kstep, ntile, lane).
// Fragment element i (i<8):  K = kb + half*8 + i ; element i+8: K = kb + 16 + half*8 + i ; col = ntile*16 + lid.
__global__ void pack_b(const float* __restrict__ W1s, _Float16* __restrict__ Bp){
    int t = blockIdx.x * blockDim.x + threadIdx.x;   // 0..767 = kstep*384 + ntile*32 + lane
    if (t >= 2*12*32) return;
    int lane  = t & 31;
    int ntile = (t >> 5) % 12;
    int kstep = t / (12*32);
    int half  = lane >> 4;
    int col   = ntile*16 + (lane & 15);
    int kb    = kstep*32;
    int k1    = kb + half*8;
    int k2    = kb + 16 + half*8;
    _Float16* fr = Bp + (size_t)t * 16;
    #pragma unroll
    for (int i = 0; i < 8; ++i){
        fr[i]     = (_Float16)W1s[(k1 + i)*F1 + col];
        fr[i + 8] = (_Float16)W1s[(k2 + i)*F1 + col];
    }
}

// ---- encoder: one wave per node; h -> f16 hh, plus folded attention scalars a_src/a_dst (f32, full precision)
__global__ void encoder_kernel(const float* __restrict__ x,
                               const float* __restrict__ Wc, const float* __restrict__ bc,
                               const float* __restrict__ Wt, const float* __restrict__ bt,
                               const float* __restrict__ u_src, const float* __restrict__ u_dst,
                               _Float16* __restrict__ hh, float* __restrict__ a_src, float* __restrict__ a_dst){
    int n = (blockIdx.x * blockDim.x + threadIdx.x) >> 5;
    int lane = threadIdx.x & 31;
    if (n >= N_NODES) return;
    const float m  = x[n*5 + 0];
    const float x1 = x[n*5 + 1], x2 = x[n*5 + 2], x3 = x[n*5 + 3];
    float as[3] = {0.f,0.f,0.f}, ad[3] = {0.f,0.f,0.f};
    #pragma unroll
    for (int rep = 0; rep < 2; ++rep){
        int c = lane + rep*32;
        float comm = x1*Wc[c] + x2*Wc[HIDDEN + c] + bc[c];
        float toll = x3*Wt[c] + bt[c];
        float hv = eluf(comm*(1.f - m) + toll*m);
        hh[(size_t)n*HIDDEN + c] = (_Float16)hv;
        #pragma unroll
        for (int h = 0; h < 3; ++h){
            as[h] += hv * u_src[c*3 + h];
            ad[h] += hv * u_dst[c*3 + h];
        }
    }
    #pragma unroll
    for (int off = 16; off; off >>= 1){
        #pragma unroll
        for (int h = 0; h < 3; ++h){
            as[h] += __shfl_xor(as[h], off, 32);
            ad[h] += __shfl_xor(ad[h], off, 32);
        }
    }
    if (lane == 0){
        #pragma unroll
        for (int h = 0; h < 3; ++h){ a_src[n*3 + h] = as[h]; a_dst[n*3 + h] = ad[h]; }
    }
}

// ---- xs = hh(N x 64, f16) @ W1 (64 x 192, pre-packed f16 fragments), f32 accumulate.
// One wave per 16x16 tile. M = 3125*16, K = 2*32, Ncols = 12*16, all exact (ISA 7.12.2 layout).
__global__ void gemm_xs_wmma(const _Float16* __restrict__ hh, const _Float16* __restrict__ Bp,
                             float* __restrict__ xs){
    const int mtile = blockIdx.x;       // 0..3124
    const int ntile = blockIdx.y;       // 0..11
    const int lane  = threadIdx.x & 31;
    const int half  = lane >> 4;        // 0: K{0..7,16..23}, 1: K{8..15,24..31}
    const int lid   = lane & 15;
    const int row   = mtile*16 + lid;   // A-matrix row (M)
    const int col   = ntile*16 + lid;   // C column (N)
    const v16h* __restrict__ Bfrag = (const v16h*)Bp;
    v8f c = {};
    #pragma unroll
    for (int kstep = 0; kstep < 2; ++kstep){
        const int kb = kstep*32;
        const int k1 = kb + half*8;
        const int k2 = kb + 16 + half*8;
        v8h alo = *(const v8h*)(hh + (size_t)row*HIDDEN + k1);
        v8h ahi = *(const v8h*)(hh + (size_t)row*HIDDEN + k2);
        v16h a = __builtin_shufflevector(alo, ahi, 0,1,2,3,4,5,6,7,8,9,10,11,12,13,14,15);
        v16h b = Bfrag[(kstep*12 + ntile)*32 + lane];
        c = __builtin_amdgcn_wmma_f32_16x16x32_f16(false, a, false, b, (short)0, c, false, false);
    }
    const int mofs = half*8;
    #pragma unroll
    for (int r = 0; r < 8; ++r)
        xs[(size_t)(mtile*16 + mofs + r)*F1 + col] = c[r];
}

// ---- layer-1 edge pass 1: raw attention + segment max (encoded-uint atomicMax)
__global__ void edge1_p1(const int* __restrict__ ei, const float* __restrict__ ea,
                         const float* __restrict__ a_src, const float* __restrict__ a_dst,
                         const float* __restrict__ ce,
                         float* __restrict__ alpha_raw, unsigned* __restrict__ amax){
    int e = blockIdx.x * blockDim.x + threadIdx.x;
    if (e >= N_EDGES) return;
    int s = ei[e], d = ei[N_EDGES + e];
    float eav = ea[e];
    #pragma unroll
    for (int h = 0; h < 3; ++h){
        float a = lrelu(a_src[s*3 + h] + a_dst[d*3 + h] + eav*ce[h]);
        alpha_raw[(size_t)e*3 + h] = a;
        atomicMax(&amax[d*3 + h], encF(a));
    }
}

// ---- layer-1 edge pass 2: w = exp(raw - max); denom += w; acc[dst] += w * xs[src]  (64 threads/edge)
__global__ void edge1_p2(const int* __restrict__ ei, const float* __restrict__ alpha_raw,
                         const unsigned* __restrict__ amax, const float* __restrict__ xs,
                         float* __restrict__ denom, float* __restrict__ acc){
    __shared__ float wsh[4][3];
    int el = threadIdx.x >> 6;          // 0..3
    int t  = threadIdx.x & 63;
    int e  = blockIdx.x*4 + el;
    bool valid = e < N_EDGES;
    int s = 0, d = 0;
    if (valid){ s = ei[e]; d = ei[N_EDGES + e]; }
    if (valid && t < 3){
        float w = expf(alpha_raw[(size_t)e*3 + t] - decF(amax[d*3 + t]));
        atomicAdd(&denom[d*3 + t], w);
        wsh[el][t] = w;
    }
    __syncthreads();
    if (valid){
        #pragma unroll
        for (int j = 0; j < 3; ++j){
            int idx = j*64 + t;         // head = j, coalesced 64-float chunks
            atomicAdd(&acc[(size_t)d*F1 + idx], wsh[el][j] * xs[(size_t)s*F1 + idx]);
        }
    }
}

// ---- layer-1 finalize (in place): h1 = elu(acc/denom + b1)
__global__ void node1_fin(float* __restrict__ acc, const float* __restrict__ denom,
                          const float* __restrict__ b1){
    int i = blockIdx.x * blockDim.x + threadIdx.x;   // < 9.6M
    if (i >= N_NODES * F1) return;
    int n = i / F1, j = i % F1, h = j >> 6;
    float v = acc[i] / (denom[n*3 + h] + 1e-16f) + b1[j];
    acc[i] = eluf(v);
}

// ---- layer-2 node scalars: xs2 = h1 . W2s ; as2 = xs2*att2s ; ad2 = (h1 . W2d)*att2d
__global__ void node2_scal(const float* __restrict__ h1, const float* __restrict__ W2s,
                           const float* __restrict__ W2d, const float* __restrict__ at2s,
                           const float* __restrict__ at2d,
                           float* __restrict__ xs2, float* __restrict__ as2, float* __restrict__ ad2){
    int n = (blockIdx.x * blockDim.x + threadIdx.x) >> 5;
    int lane = threadIdx.x & 31;
    if (n >= N_NODES) return;
    float s1 = 0.f, s2 = 0.f;
    for (int j = lane; j < F1; j += 32){
        float hv = h1[(size_t)n*F1 + j];
        s1 += hv * W2s[j];
        s2 += hv * W2d[j];
    }
    #pragma unroll
    for (int off = 16; off; off >>= 1){
        s1 += __shfl_xor(s1, off, 32);
        s2 += __shfl_xor(s2, off, 32);
    }
    if (lane == 0){
        xs2[n] = s1;
        as2[n] = s1 * at2s[0];
        ad2[n] = s2 * at2d[0];
    }
}

__global__ void edge2_p1(const int* __restrict__ ei, const float* __restrict__ ea,
                         const float* __restrict__ as2, const float* __restrict__ ad2,
                         const float* __restrict__ W2e, const float* __restrict__ at2e,
                         float* __restrict__ raw2, unsigned* __restrict__ amax2){
    int e = blockIdx.x * blockDim.x + threadIdx.x;
    if (e >= N_EDGES) return;
    int s = ei[e], d = ei[N_EDGES + e];
    float a = lrelu(as2[s] + ad2[d] + ea[e] * W2e[0] * at2e[0]);
    raw2[e] = a;
    atomicMax(&amax2[d], encF(a));
}

__global__ void edge2_p2(const int* __restrict__ ei, const float* __restrict__ raw2,
                         const unsigned* __restrict__ amax2, const float* __restrict__ xs2,
                         float* __restrict__ den2, float* __restrict__ num2){
    int e = blockIdx.x * blockDim.x + threadIdx.x;
    if (e >= N_EDGES) return;
    int s = ei[e], d = ei[N_EDGES + e];
    float w = expf(raw2[e] - decF(amax2[d]));
    atomicAdd(&den2[d], w);
    atomicAdd(&num2[d], w * xs2[s]);
}

__global__ void final_out(const float* __restrict__ num2, const float* __restrict__ den2,
                          const float* __restrict__ b2, const float* __restrict__ x,
                          float* __restrict__ out){
    int n = blockIdx.x * blockDim.x + threadIdx.x;
    if (n >= N_NODES) return;
    out[n] = (num2[n] / (den2[n] + 1e-16f) + b2[0]) * x[n*5];   // * mask
}

extern "C" void kernel_launch(void* const* d_in, const int* in_sizes, int n_in,
                              void* d_out, int out_size, void* d_ws, size_t ws_size,
                              hipStream_t stream){
    (void)in_sizes; (void)n_in; (void)out_size; (void)ws_size;
    const float* x    = (const float*)d_in[0];
    const int*   ei   = (const int*)d_in[1];
    const float* ea   = (const float*)d_in[2];
    const float* Wc   = (const float*)d_in[3];
    const float* bc   = (const float*)d_in[4];
    const float* Wt   = (const float*)d_in[5];
    const float* bt   = (const float*)d_in[6];
    const float* W1s  = (const float*)d_in[7];
    const float* W1d  = (const float*)d_in[8];
    const float* a1s  = (const float*)d_in[9];
    const float* a1d  = (const float*)d_in[10];
    const float* W1e  = (const float*)d_in[11];
    const float* a1e  = (const float*)d_in[12];
    const float* b1   = (const float*)d_in[13];
    const float* W2s  = (const float*)d_in[14];
    const float* W2d  = (const float*)d_in[15];
    const float* at2s = (const float*)d_in[16];
    const float* at2d = (const float*)d_in[17];
    const float* W2e  = (const float*)d_in[18];
    const float* at2e = (const float*)d_in[19];
    const float* b2   = (const float*)d_in[20];
    float* out = (float*)d_out;

    char* p = (char*)d_ws;
    auto carve = [&](size_t bytes) -> char* {
        char* r = p; p += (bytes + 255) & ~(size_t)255; return r;
    };
    _Float16* hh    = (_Float16*)carve((size_t)N_NODES*HIDDEN*2);
    _Float16* Bp    = (_Float16*)carve((size_t)2*12*32*16*2);    // packed B fragments
    float* u_src    = (float*)carve(HIDDEN*3*4);
    float* u_dst    = (float*)carve(HIDDEN*3*4);
    float* ce       = (float*)carve(3*4);
    float* a_srcB   = (float*)carve((size_t)N_NODES*3*4);
    float* a_dstB   = (float*)carve((size_t)N_NODES*3*4);
    float* xs       = (float*)carve((size_t)N_NODES*F1*4);
    float* acc      = (float*)carve((size_t)N_NODES*F1*4);       // becomes h1 in place
    float* alpha    = (float*)carve((size_t)N_EDGES*3*4);
    unsigned* amax  = (unsigned*)carve((size_t)N_NODES*3*4);
    float* denom    = (float*)carve((size_t)N_NODES*3*4);
    float* xs2      = (float*)carve((size_t)N_NODES*4);
    float* as2      = (float*)carve((size_t)N_NODES*4);
    float* ad2      = (float*)carve((size_t)N_NODES*4);
    float* raw2     = (float*)carve((size_t)N_EDGES*4);
    unsigned* amax2 = (unsigned*)carve((size_t)N_NODES*4);
    float* den2     = (float*)carve((size_t)N_NODES*4);
    float* num2     = (float*)carve((size_t)N_NODES*4);

    // re-init accumulators every call (deterministic)
    hipMemsetAsync(amax,  0, (size_t)N_NODES*3*4,  stream);
    hipMemsetAsync(denom, 0, (size_t)N_NODES*3*4,  stream);
    hipMemsetAsync(acc,   0, (size_t)N_NODES*F1*4, stream);
    hipMemsetAsync(amax2, 0, (size_t)N_NODES*4,    stream);
    hipMemsetAsync(den2,  0, (size_t)N_NODES*4,    stream);
    hipMemsetAsync(num2,  0, (size_t)N_NODES*4,    stream);

    fold_kernel<<<1, 192, 0, stream>>>(W1s, W1d, a1s, a1d, W1e, a1e, u_src, u_dst, ce);
    pack_b<<<3, 256, 0, stream>>>(W1s, Bp);
    encoder_kernel<<<(N_NODES + 7)/8, 256, 0, stream>>>(x, Wc, bc, Wt, bt, u_src, u_dst, hh, a_srcB, a_dstB);
    gemm_xs_wmma<<<dim3(N_NODES/16, F1/16), 32, 0, stream>>>(hh, Bp, xs);
    edge1_p1<<<(N_EDGES + 255)/256, 256, 0, stream>>>(ei, ea, a_srcB, a_dstB, ce, alpha, amax);
    edge1_p2<<<(N_EDGES + 3)/4, 256, 0, stream>>>(ei, alpha, amax, xs, denom, acc);
    node1_fin<<<(N_NODES*F1 + 255)/256, 256, 0, stream>>>(acc, denom, b1);
    node2_scal<<<(N_NODES + 7)/8, 256, 0, stream>>>(acc, W2s, W2d, at2s, at2d, xs2, as2, ad2);
    edge2_p1<<<(N_EDGES + 255)/256, 256, 0, stream>>>(ei, ea, as2, ad2, W2e, at2e, raw2, amax2);
    edge2_p2<<<(N_EDGES + 255)/256, 256, 0, stream>>>(ei, raw2, amax2, xs2, den2, num2);
    final_out<<<(N_NODES + 255)/256, 256, 0, stream>>>(num2, den2, b2, x, out);
}
